// LinearAttention_63582695850854
// MI455X (gfx1250) — compile-verified, matrix operand order checked by reading, not run on hardware
//
#include <hip/hip_runtime.h>
#include <hip/hip_bf16.h>

// ---------------------------------------------------------------------------
// LinearAttention for MI455X (gfx1250), wave32.
//  - all GEMMs: v_wmma_f32_16x16x32_bf16, fp32 accumulation
//  - all intermediates + weights pre-converted/pre-transposed to bf16 so both
//    A and B tiles stage via global_load_async_to_lds_b128 (ASYNCcnt),
//    LDS double-buffered; fragments via ds_load_b128
//  - GM/GN template flags compile out bounds guards for tile-aligned dims so
//    the dominant GEMMs get straight-line async staging (no exec-mask ladders)
// ---------------------------------------------------------------------------

typedef __attribute__((ext_vector_type(16))) __bf16 v16bf;
typedef __attribute__((ext_vector_type(8)))  __bf16 v8bf;
typedef __attribute__((ext_vector_type(8)))  float  v8f;

union FragAB { v16bf v; v8bf h[2]; };

__device__ __forceinline__ __bf16 f2bf(float f) {
  unsigned u = __builtin_bit_cast(unsigned, f);
  unsigned r = u + 0x7FFFu + ((u >> 16) & 1u);   // round-to-nearest-even
  return __builtin_bit_cast(__bf16, (unsigned short)(r >> 16));
}

// Async global->LDS 16-byte copy, tracked by ASYNCcnt.
__device__ __forceinline__ void async_copy_b128(const void* gptr, void* lptr) {
  unsigned lds = (unsigned)(unsigned long long)lptr;  // low 32 bits = LDS offset
  asm volatile("global_load_async_to_lds_b128 %0, %1, off"
               :: "v"(lds), "v"(gptr) : "memory");
}
__device__ __forceinline__ void wait_asynccnt0() {
  asm volatile("s_wait_asynccnt 0x0" ::: "memory");
}

constexpr int BM = 128, BN = 128, BK = 32;

// C = A * B (+bias)(relu)(row-scale 1/(norm+eps)).  A,B bf16; C fp32 or bf16.
// A: MxK row-major.  B: BT ? [N][K] (pre-transposed, async path) : [K][N].
// OUT_TRANS: store C[col*ldc+row].  GM/GN: emit M/N bounds guards.
// Batch z: off = (z/zmod)*sXh + (z%zmod)*sXl.
template <bool BT, bool HAS_BIAS, bool RELU, bool ROWSCALE, bool OUT_BF16,
          bool OUT_TRANS, bool GM, bool GN>
__global__ __launch_bounds__(256, 2) void gemm_bf16_wmma(
    const __bf16* __restrict__ A, const __bf16* __restrict__ B,
    void* __restrict__ Cv, const float* __restrict__ bias,
    const float* __restrict__ rowscale,
    int M, int N, int K, int lda, int ldb, int ldc,
    long sAh, long sAl, long sBh, long sBl, long sCh, long sCl,
    long sSh, long sSl, int zmod)
{
  __shared__ __align__(16) __bf16 Als[2][BM][BK];   // [m][k]
  __shared__ __align__(16) __bf16 Bls[2][BN][BK];   // [n][k]

  const int tid  = threadIdx.x;
  const int lane = tid & 31;
  const int wave = tid >> 5;
  const int wm = (wave & 3) * 32;
  const int wn = (wave >> 2) * 64;
  const int lr = lane & 15;
  const int lh = lane >> 4;

  const int z  = blockIdx.z;
  const int zb = z / zmod, zl = z % zmod;
  A += zb * sAh + zl * sAl;
  B += zb * sBh + zl * sBl;
  const float* rs = rowscale + zb * sSh + zl * sSl;

  const int m0 = blockIdx.y * BM;
  const int n0 = blockIdx.x * BN;

  auto stageA = [&](int buf, int k0) {      // pure async DMA copy
#pragma unroll
    for (int j = 0; j < 2; ++j) {
      const int flat = tid + j * 256;       // 16B chunk id (0..511)
      const int row  = flat >> 2;           // 4 chunks per 64B row
      const int ce   = (flat & 3) * 8;
      if (!GM || m0 + row < M)
        async_copy_b128(A + (long)(m0 + row) * lda + k0 + ce,
                        &Als[buf][row][ce]);
    }
  };
  auto stageB = [&](int buf, int k0) {
    if constexpr (BT) {                     // pre-transposed: async DMA copy
#pragma unroll
      for (int j = 0; j < 2; ++j) {
        const int flat = tid + j * 256;
        const int row  = flat >> 2;
        const int ce   = (flat & 3) * 8;
        if (!GN || n0 + row < N)
          async_copy_b128(B + (long)(n0 + row) * ldb + k0 + ce,
                          &Bls[buf][row][ce]);
      }
    } else {                                // [K][N]: load + transpose scatter
#pragma unroll
      for (int j = 0; j < 2; ++j) {
        const int flat8 = tid + j * 256;    // 0..511
        const int kk    = flat8 >> 4;
        const int n     = (flat8 & 15) * 8;
        if (!GN || n0 + n < N) {
          v8bf v = *(const v8bf*)(B + (long)(k0 + kk) * ldb + n0 + n);
#pragma unroll
          for (int t = 0; t < 8; ++t) Bls[buf][n + t][kk] = v[t];
        }
      }
    }
  };

  v8f acc[2][4] = {};

  const int nk = K / BK;
  stageA(0, 0);
  stageB(0, 0);
  wait_asynccnt0();
  __syncthreads();

  for (int t = 0; t < nk; ++t) {
    const int cur = t & 1;
    if (t + 1 < nk) {                       // overlap next-tile DMA with WMMAs
      stageA(cur ^ 1, (t + 1) * BK);
      stageB(cur ^ 1, (t + 1) * BK);
      if (t + 2 < nk)
        __builtin_prefetch(A + (long)(m0 + (tid & 127)) * lda + (t + 2) * BK, 0, 0);
    }

    // fragments per CDNA5 16-bit layout:
    // lane<16: elems 0..7 = K 0..7, 8..15 = K 16..23 ; lane>=16: +8
    FragAB af[2], bf[4];
#pragma unroll
    for (int mt = 0; mt < 2; ++mt) {
      const int r = wm + mt * 16 + lr;
      af[mt].h[0] = *(const v8bf*)&Als[cur][r][lh * 8];
      af[mt].h[1] = *(const v8bf*)&Als[cur][r][lh * 8 + 16];
    }
#pragma unroll
    for (int nt = 0; nt < 4; ++nt) {
      const int c = wn + nt * 16 + lr;
      bf[nt].h[0] = *(const v8bf*)&Bls[cur][c][lh * 8];
      bf[nt].h[1] = *(const v8bf*)&Bls[cur][c][lh * 8 + 16];
    }
#pragma unroll
    for (int mt = 0; mt < 2; ++mt)
#pragma unroll
      for (int nt = 0; nt < 4; ++nt)
        acc[mt][nt] = __builtin_amdgcn_wmma_f32_16x16x32_bf16(
            false, af[mt].v, false, bf[nt].v, (short)0, acc[mt][nt],
            false, false);

    wait_asynccnt0();
    __syncthreads();
  }

  // ---- epilogue ----
  const int base_row = m0 + wm;
  const int base_col = n0 + wn;
#pragma unroll
  for (int mt = 0; mt < 2; ++mt) {
#pragma unroll
    for (int r = 0; r < 8; ++r) {
      const int row = base_row + mt * 16 + lh * 8 + r;
      if (GM && row >= M) continue;
      float scale = 1.0f;
      if constexpr (ROWSCALE) scale = 1.0f / (rs[row] + 1e-6f);
#pragma unroll
      for (int nt = 0; nt < 4; ++nt) {
        const int col = base_col + nt * 16 + lr;
        if (GN && col >= N) continue;
        float v = acc[mt][nt][r];
        if constexpr (HAS_BIAS) v += bias[col];
        if constexpr (RELU) v = v > 0.f ? v : 0.f;
        if constexpr (ROWSCALE) v *= scale;
        const long cidx = OUT_TRANS ? ((long)col * ldc + row)
                                    : ((long)row * ldc + col);
        if constexpr (OUT_BF16)
          ((__bf16*)Cv)[zb * sCh + zl * sCl + cidx] = f2bf(v);
        else
          ((float*)Cv)[zb * sCh + zl * sCl + cidx] = v;
      }
    }
  }
}

// fp32 -> bf16 bulk convert (8 elems/thread)
__global__ __launch_bounds__(256) void f32_to_bf16_kernel(
    const float* __restrict__ in, __bf16* __restrict__ out, long n)
{
  const long i = ((long)blockIdx.x * 256 + threadIdx.x) * 8;
  if (i >= n) return;
  v8f f = *(const v8f*)(in + i);
  *(v8bf*)(out + i) = __builtin_convertvector(f, v8bf);
}

// fp32 [R][C] -> bf16 transposed [C][R] (32x32 LDS tiles; R,C multiples of 32)
__global__ __launch_bounds__(256) void f32_to_bf16_T_kernel(
    const float* __restrict__ in, __bf16* __restrict__ out, int R, int C)
{
  __shared__ float tile[32][33];
  const int r0 = blockIdx.y * 32, c0 = blockIdx.x * 32;
  const int tc = threadIdx.x & 31, t8 = threadIdx.x >> 5;
#pragma unroll
  for (int j = 0; j < 4; ++j) {
    const int r = t8 + j * 8;
    tile[r][tc] = in[(long)(r0 + r) * C + c0 + tc];
  }
  __syncthreads();
#pragma unroll
  for (int j = 0; j < 4; ++j) {
    const int c = t8 + j * 8;
    out[(long)(c0 + c) * R + r0 + tc] = f2bf(tile[tc][c]);
  }
}

// ksum[z][d] = sum_n kprimeT[z][d][n]  (contiguous row reduction over Nseq)
__global__ __launch_bounds__(256) void ksum_kernel(
    const __bf16* __restrict__ kpT, float* __restrict__ ksum, int Nseq)
{
  __shared__ float part[4][64];
  const int z = blockIdx.x;
  const int d = threadIdx.x >> 2;
  const int c = threadIdx.x & 3;
  const int per = Nseq >> 2;
  const __bf16* row = kpT + ((long)z * 64 + d) * Nseq + c * per;
  float s = 0.f;
  for (int i = 0; i < per; i += 8) {
    v8bf v = *(const v8bf*)(row + i);
#pragma unroll
    for (int t = 0; t < 8; ++t) s += (float)v[t];
  }
  part[c][d] = s;
  __syncthreads();
  if (threadIdx.x < 64) {
    const int dd = threadIdx.x;
    ksum[z * 64 + dd] = part[0][dd] + part[1][dd] + part[2][dd] + part[3][dd];
  }
}

// norm[z][n] = dot(qprime[z][n][:], ksum[z][:])
__global__ __launch_bounds__(256) void norm_kernel(
    const __bf16* __restrict__ qp, const float* __restrict__ ksum,
    float* __restrict__ nrm, int Nseq)
{
  __shared__ float ks[64];
  const int blocksPerZ = Nseq / 256;
  const int z  = blockIdx.x / blocksPerZ;
  const int nb = blockIdx.x % blocksPerZ;
  if (threadIdx.x < 64) ks[threadIdx.x] = ksum[z * 64 + threadIdx.x];
  __syncthreads();
  const int n = nb * 256 + threadIdx.x;
  const __bf16* row = qp + ((long)z * Nseq + n) * 64;
  float s = 0.f;
#pragma unroll
  for (int d = 0; d < 64; ++d) s += (float)row[d] * ks[d];
  nrm[(long)z * Nseq + n] = s;
}

// ---------------------------------------------------------------------------
extern "C" void kernel_launch(void* const* d_in, const int* in_sizes, int n_in,
                              void* d_out, int out_size, void* d_ws, size_t ws_size,
                              hipStream_t stream)
{
  const float* x     = (const float*)d_in[0];  // [4,4096,1024]
  const float* W_qkv = (const float*)d_in[1];  // [1024,3072]
  const float* W_f   = (const float*)d_in[2];  // [64,64]
  const float* b_f   = (const float*)d_in[3];  // [64]
  const float* W_out = (const float*)d_in[4];  // [1024,1024]
  const float* b_out = (const float*)d_in[5];  // [1024]
  float* out = (float*)d_out;                  // [4,4096,1024] fp32

  constexpr int  Bb = 4, Nseq = 4096, Cdim = 1024, H = 16, D = 64;
  constexpr int  Mrows = Bb * Nseq;               // 16384
  constexpr long XN   = (long)Mrows * Cdim;       // 16,777,216
  constexpr long QKV  = (long)Mrows * 3 * Cdim;   // 50,331,648
  constexpr long PERZ = (long)Nseq * D;           // 262,144

  __bf16* wsb    = (__bf16*)d_ws;
  __bf16* xbf    = wsb;                           // [16384][1024]
  __bf16* wqkvT  = xbf   + XN;                    // [3072][1024]
  __bf16* wfT    = wqkvT + (long)Cdim * 3 * Cdim; // [64][64]
  __bf16* woutT  = wfT   + D * D;                 // [1024][1024]
  __bf16* qkvb   = woutT + (long)Cdim * Cdim;     // [16384][3072]
  __bf16* qpb    = qkvb  + QKV;                   // [64][4096][64]
  __bf16* kpT    = qpb   + 64 * PERZ;             // [64][64][4096]
  __bf16* kvT    = kpT   + 64 * PERZ;             // [64][64][64]  (kv^T)
  __bf16* attnb  = kvT   + 64L * D * D;           // [16384][1024]
  float*  ksum   = (float*)(attnb + XN);          // [64][64]
  float*  nrm    = ksum + 64 * D;                 // [64][4096]

  const dim3 blk(256);
  // 0) one-time fp32->bf16 conversions; B operands transposed to [N][K]
  f32_to_bf16_kernel<<<dim3((unsigned)(XN / 8 / 256)), blk, 0, stream>>>(x, xbf, XN);
  f32_to_bf16_T_kernel<<<dim3(3 * Cdim / 32, Cdim / 32), blk, 0, stream>>>(
      W_qkv, wqkvT, Cdim, 3 * Cdim);
  f32_to_bf16_T_kernel<<<dim3(D / 32, D / 32), blk, 0, stream>>>(W_f, wfT, D, D);
  f32_to_bf16_T_kernel<<<dim3(Cdim / 32, Cdim / 32), blk, 0, stream>>>(
      W_out, woutT, Cdim, Cdim);

  const long sQKVb = (long)Nseq * 3 * Cdim;   // batch-b stride inside qkv
  const long sPZh  = PERZ;
  const long sPZb  = 16 * PERZ;

  // 1) qkv = x @ W_qkv  (bf16 out; fully tile-aligned -> no guards)
  gemm_bf16_wmma<true, false, false, false, true, false, false, false>
      <<<dim3(3 * Cdim / BN, Mrows / BM, 1), blk, 0, stream>>>(
      xbf, wqkvT, qkvb, nullptr, nrm,
      Mrows, 3 * Cdim, Cdim, Cdim, Cdim, 3 * Cdim,
      0, 0, 0, 0, 0, 0, 0, 0, 1);

  // 2) q' = relu(q @ W_f + b_f)  -> qpb [z][n][d]   (N=64 -> GN only)
  gemm_bf16_wmma<true, true, true, false, true, false, false, true>
      <<<dim3(1, Nseq / BM, Bb * H), blk, 0, stream>>>(
      qkvb, wfT, qpb, b_f, nrm,
      Nseq, D, D, 3 * Cdim, D, D,
      sQKVb, 64, 0, 0, sPZb, sPZh, 0, 0, H);

  // 3) k' = relu(k @ W_f + b_f)  -> stored transposed kpT [z][d][n]
  gemm_bf16_wmma<true, true, true, false, true, true, false, true>
      <<<dim3(1, Nseq / BM, Bb * H), blk, 0, stream>>>(
      qkvb + Cdim, wfT, kpT, b_f, nrm,
      Nseq, D, D, 3 * Cdim, D, Nseq,
      sQKVb, 64, 0, 0, sPZb, sPZh, 0, 0, H);

  // 4) kv^T = (k'^T @ v)^T : A = kpT (row-major), B = v slice [K][N] scatter
  gemm_bf16_wmma<false, false, false, false, true, true, true, true>
      <<<dim3(1, 1, Bb * H), blk, 0, stream>>>(
      kpT, qkvb + 2 * Cdim, kvT, nullptr, nrm,
      D, D, Nseq, Nseq, 3 * Cdim, D,
      sPZb, sPZh, sQKVb, 64, 16L * D * D, (long)D * D, 0, 0, H);

  // 5) k_sum (contiguous rows of kpT), then normalizer = q' . k_sum
  ksum_kernel<<<dim3(Bb * H), blk, 0, stream>>>(kpT, ksum, Nseq);
  norm_kernel<<<dim3(Bb * H * (Nseq / 256)), blk, 0, stream>>>(qpb, ksum, nrm, Nseq);

  // 6) attn = (q' @ kv) / (norm + 1e-6) -> [B,N,H*D] bf16 ; B = kvT (BT path)
  gemm_bf16_wmma<true, false, false, true, true, false, false, true>
      <<<dim3(1, Nseq / BM, Bb * H), blk, 0, stream>>>(
      qpb, kvT, attnb, nullptr, nrm,
      Nseq, D, D, D, D, Cdim,
      sPZb, sPZh, 16L * D * D, (long)D * D,
      (long)Nseq * Cdim, 64, 16L * Nseq, Nseq, H);

  // 7) out = attn @ W_out + b_out  (fp32 out; fully tile-aligned -> no guards)
  gemm_bf16_wmma<true, true, false, false, false, false, false, false>
      <<<dim3(Cdim / BN, Mrows / BM, 1), blk, 0, stream>>>(
      attnb, woutT, out, b_out, nrm,
      Mrows, Cdim, Cdim, Cdim, Cdim, Cdim,
      0, 0, 0, 0, 0, 0, 0, 0, 1);
}
